// Encoder_8718783611631
// MI455X (gfx1250) — matrix operand order repaired and checked by reading, not compiled, not run on hardware
//
#include <hip/hip_runtime.h>
#include <hip/hip_bf16.h>

// ---------------------------------------------------------------------------
// MI455X (gfx1250) transformer encoder, bf16 WMMA path.
// Async global->LDS staging (GLOBAL_LOAD_ASYNC_TO_LDS_B128 / ASYNCcnt),
// double-buffered LDS, one barrier per K-step, 32x64 wave tiles
// (8 v_wmma_f32_16x16x32_bf16 per K-step per wave). Epilogue variants are
// fully templated (GELU / bf16-out / residual) so no runtime branches remain
// in the unrolled store loops.
// ---------------------------------------------------------------------------

typedef __bf16 bf16;
typedef __attribute__((ext_vector_type(16))) __bf16 v16bf;
typedef __attribute__((ext_vector_type(8)))  __bf16 v8bf;
typedef __attribute__((ext_vector_type(8)))  float  v8f;
typedef __attribute__((ext_vector_type(4)))  int    v4i;

union Frag16 { v16bf v; v8bf h[2]; };

constexpr int CL = 12, CD = 1024, CNH = 16, CDFF = 4096, CB = 512, CS = 17, CDK = 64;
constexpr float CRW = 0.2f;
constexpr float CEPS = 1e-5f;

__device__ __forceinline__ float geluf(float x) {
  return 0.5f * x * (1.0f + erff(x * 0.70710678118654752440f));
}

// --------------------------- async global -> LDS ---------------------------
#if __has_builtin(__builtin_amdgcn_global_load_async_to_lds_b128)
#define HAS_ASYNC_LDS 1
#else
#define HAS_ASYNC_LDS 0
#endif

__device__ __forceinline__ void stage16(const bf16* g, bf16* l) {
#if HAS_ASYNC_LDS
  // (int4 AS1*, int4 AS3*, imm offset, imm cpol)
  __builtin_amdgcn_global_load_async_to_lds_b128(
      (__attribute__((address_space(1))) v4i*)(uintptr_t)g,
      (__attribute__((address_space(3))) v4i*)(uintptr_t)l, 0, 0);
#else
  *(v8bf*)l = *(const v8bf*)g;
#endif
}
__device__ __forceinline__ void stage_wait() {
#if HAS_ASYNC_LDS
#if __has_builtin(__builtin_amdgcn_s_wait_asynccnt)
  __builtin_amdgcn_s_wait_asynccnt(0);
#else
  asm volatile("s_wait_asynccnt 0x0" ::: "memory");
#endif
#endif
}

// ---------------------------------------------------------------------------
// GEMM: C[M][N] = A_bf16[M][K] * Bt_bf16[N][K]^T + bias (+gelu) (+residual)
// Block tile 128x128, BK=32. 8 wave32 in a 4(M) x 2(N) grid; each wave owns a
// 32x64 macro-tile = 2x4 v_wmma_f32_16x16x32_bf16 accumulators.
// A and B tiles are both 128 rows x 32 bf16 -> identical, divergence-free
// staging: each of the 256 threads moves 32B of A and 32B of B per K-step.
// ---------------------------------------------------------------------------
constexpr int BM = 128, BN = 128, BK = 32, LDP = BK + 8;

template<bool GELU, bool OUTBF, bool RES>
__global__ __launch_bounds__(256) void k_gemm(const bf16* __restrict__ A,
                                              const bf16* __restrict__ Bt,
                                              const float* __restrict__ bias,
                                              const float* __restrict__ res,
                                              void* __restrict__ outp,
                                              int M, int N, int K) {
  __shared__ bf16 As[2][BM][LDP];
  __shared__ bf16 Bs[2][BN][LDP];
  const int tid = threadIdx.x;
  const int m0 = blockIdx.x * BM, n0 = blockIdx.y * BN;
  const int wid = tid >> 5, lane = tid & 31;
  const int wm = wid & 3, wn = wid >> 2;          // wave grid 4x2
  const int lr = lane & 15, lh = lane >> 4;       // lane row / lane half
  v8f acc[2][4] = {};

  const int row = tid >> 1;                       // 128 rows, 2 threads/row
  const int seg = (tid & 1) * 16;                 // 16 bf16 per thread
  const bf16* Ap = A + (size_t)(m0 + row) * K + seg;
  const bf16* Bp = Bt + (size_t)(n0 + row) * K + seg;

  auto stage_tile = [&](int buf, int k0) {
    stage16(Ap + k0,     &As[buf][row][seg]);
    stage16(Ap + k0 + 8, &As[buf][row][seg + 8]);
    stage16(Bp + k0,     &Bs[buf][row][seg]);
    stage16(Bp + k0 + 8, &Bs[buf][row][seg + 8]);
  };

  stage_tile(0, 0);
  stage_wait();
  __syncthreads();

  int buf = 0;
  for (int k0 = 0; k0 < K; k0 += BK) {
    const bool more = (k0 + BK) < K;
    if (more) stage_tile(buf ^ 1, k0 + BK);       // overlap next tile (ASYNCcnt)
    if (k0 + 2 * BK < K) {
      __builtin_prefetch(Ap + k0 + 2 * BK, 0, 1); // global_prefetch_b8 (L2 warm)
      __builtin_prefetch(Bp + k0 + 2 * BK, 0, 1);
    }

    // A frag (16x32 bf16): lane half selects K base 0/8; chunks [bk,bk+8) and
    // [bk+16,bk+24). B frag (32x16): same K chunks as A.
    Frag16 af[2], bfr[4];
#pragma unroll
    for (int im = 0; im < 2; ++im) {
      const int r = wm * 32 + im * 16 + lr;
      const int bk = lh * 8;
      af[im].h[0] = *(const v8bf*)&As[buf][r][bk];
      af[im].h[1] = *(const v8bf*)&As[buf][r][bk + 16];
    }
#pragma unroll
    for (int in = 0; in < 4; ++in) {
      const int c = wn * 64 + in * 16 + lr;
      bfr[in].h[0] = *(const v8bf*)&Bs[buf][c][lh * 8];
      bfr[in].h[1] = *(const v8bf*)&Bs[buf][c][lh * 8 + 16];
    }
#pragma unroll
    for (int im = 0; im < 2; ++im)
#pragma unroll
      for (int in = 0; in < 4; ++in)
        acc[im][in] = __builtin_amdgcn_wmma_f32_16x16x32_bf16(
            false, af[im].v, false, bfr[in].v, (short)0, acc[im][in], false, false);

    if (more) {
      stage_wait();        // our async writes to buf^1 are in LDS
      __syncthreads();     // everyone done reading buf & writing buf^1
      buf ^= 1;
    }
  }

  // C/D layout: VGPR r -> row = base + (lane>>4)*8 + r, col = base + lane&15
#pragma unroll
  for (int im = 0; im < 2; ++im) {
#pragma unroll
    for (int in = 0; in < 4; ++in) {
      const int col = n0 + wn * 64 + in * 16 + lr;
      const float bv = bias[col];
#pragma unroll
      for (int r2 = 0; r2 < 8; ++r2) {
        const int orow = m0 + wm * 32 + im * 16 + lh * 8 + r2;
        float v = acc[im][in][r2] + bv;
        if (GELU) v = geluf(v);
        if (RES)  v += res[(size_t)orow * N + col];
        if (OUTBF) ((bf16*)outp)[(size_t)orow * N + col] = (bf16)v;
        else       ((float*)outp)[(size_t)orow * N + col] = v;
      }
    }
  }
}

// ---------------------------------------------------------------------------
// Weight convert: W f32 [K][N] -> WT bf16 [N][K] (transposed for B operand)
// ---------------------------------------------------------------------------
__global__ __launch_bounds__(256) void k_wcvt(const float* __restrict__ W,
                                              bf16* __restrict__ WT,
                                              int K, int N) {
  __shared__ float t[32][33];
  const int tx = threadIdx.x & 31, ty = threadIdx.x >> 5;
  const int n0 = blockIdx.x * 32, k0 = blockIdx.y * 32;
#pragma unroll
  for (int r = 0; r < 4; ++r)
    t[ty + r * 8][tx] = W[(size_t)(k0 + ty + r * 8) * N + n0 + tx];
  __syncthreads();
#pragma unroll
  for (int r = 0; r < 4; ++r)
    WT[(size_t)(n0 + ty + r * 8) * K + k0 + tx] = (bf16)t[tx][ty + r * 8];
}

// ---------------------------------------------------------------------------
// LayerNorm over D=1024, one row per block (256 threads, 4 elems/thread)
// ---------------------------------------------------------------------------
__global__ __launch_bounds__(256) void k_ln(const float* __restrict__ x,
                                            const float* __restrict__ g,
                                            const float* __restrict__ b,
                                            float* __restrict__ o) {
  __shared__ float red[8];
  __shared__ float bcast;
  const size_t row = blockIdx.x;
  const float* xr = x + row * (size_t)CD;
  const int tid = threadIdx.x;
  float v[4];
  float s = 0.f;
#pragma unroll
  for (int r = 0; r < 4; ++r) { v[r] = xr[tid + r * 256]; s += v[r]; }
#pragma unroll
  for (int of = 16; of; of >>= 1) s += __shfl_xor(s, of, 32);
  if ((tid & 31) == 0) red[tid >> 5] = s;
  __syncthreads();
  if (tid == 0) { float t = 0.f; for (int i = 0; i < 8; ++i) t += red[i]; bcast = t; }
  __syncthreads();
  const float mu = bcast * (1.f / CD);
  float q = 0.f;
#pragma unroll
  for (int r = 0; r < 4; ++r) { const float d = v[r] - mu; q += d * d; }
  __syncthreads();
#pragma unroll
  for (int of = 16; of; of >>= 1) q += __shfl_xor(q, of, 32);
  if ((tid & 31) == 0) red[tid >> 5] = q;
  __syncthreads();
  if (tid == 0) { float t = 0.f; for (int i = 0; i < 8; ++i) t += red[i]; bcast = t; }
  __syncthreads();
  const float inv = rsqrtf(bcast * (1.f / CD) + CEPS);
#pragma unroll
  for (int r = 0; r < 4; ++r) {
    const int c = tid + r * 256;
    o[row * CD + c] = (v[r] - mu) * inv * g[c] + b[c];
  }
}

// ---------------------------------------------------------------------------
// Attention core: per (b,h), S=17, dk=64, fp32 (tiny fraction of FLOPs)
// ---------------------------------------------------------------------------
__global__ __launch_bounds__(64) void k_attn(const float* __restrict__ Q,
                                             const float* __restrict__ Kt,
                                             const float* __restrict__ V,
                                             const int* __restrict__ mask,
                                             float* __restrict__ O) {
  const int b = blockIdx.x / CNH, h = blockIdx.x % CNH;
  __shared__ float Qs[CS][CDK], Ks[CS][CDK], Vs[CS][CDK];
  const size_t base = (size_t)b * CS * CD + (size_t)h * CDK;
  for (int idx = threadIdx.x; idx < CS * CDK; idx += blockDim.x) {
    const int s = idx / CDK, d = idx % CDK;
    Qs[s][d] = Q[base + (size_t)s * CD + d];
    Ks[s][d] = Kt[base + (size_t)s * CD + d];
    Vs[s][d] = V[base + (size_t)s * CD + d];
  }
  __syncthreads();
  const int i = threadIdx.x;
  if (i < CS) {
    float sc[CS];
    float mx = -1e30f;
    for (int j = 0; j < CS; ++j) {
      float d = 0.f;
      for (int k = 0; k < CDK; ++k) d += Qs[i][k] * Ks[j][k];
      d *= 0.125f;  // 1/sqrt(64)
      if (mask[((size_t)b * CS + i) * CS + j] == 0) d = -1e9f;
      sc[j] = d;
      if (d > mx) mx = d;
    }
    float se = 0.f;
    for (int j = 0; j < CS; ++j) { sc[j] = expf(sc[j] - mx); se += sc[j]; }
    const float inv = 1.f / se;
    for (int d = 0; d < CDK; ++d) {
      float o = 0.f;
      for (int j = 0; j < CS; ++j) o += sc[j] * Vs[j][d];
      O[base + (size_t)i * CD + d] = o * inv;
    }
  }
}

// ---------------------------------------------------------------------------
// Elementwise helpers
// ---------------------------------------------------------------------------
__global__ void k_copy(float* __restrict__ o, const float* __restrict__ x, size_t n) {
  const size_t i = (size_t)blockIdx.x * blockDim.x + threadIdx.x;
  if (i < n) o[i] = x[i];
}
__global__ void k_cvt(bf16* __restrict__ o, const float* __restrict__ x, size_t n) {
  const size_t i = (size_t)blockIdx.x * blockDim.x + threadIdx.x;
  if (i < n) o[i] = (bf16)x[i];
}
// Hin[b,j,:] = inp[b,2j+2,:] - inp[b,2j+1,:]  (bf16 out), j=0..7
__global__ void k_build_hin(bf16* __restrict__ o, const float* __restrict__ inp) {
  const size_t i = (size_t)blockIdx.x * blockDim.x + threadIdx.x;
  if (i >= (size_t)CB * 8 * CD) return;
  const size_t b = i / ((size_t)8 * CD), r = i % ((size_t)8 * CD);
  const size_t j = r / CD, d = r % CD;
  o[i] = (bf16)(inp[((size_t)b * CS + 2 * j + 2) * CD + d] -
                inp[((size_t)b * CS + 2 * j + 1) * CD + d]);
}
// Tin[b,j,:] = inp[b,2j,:] + inp[b,2j+1,:]
__global__ void k_build_tin(bf16* __restrict__ o, const float* __restrict__ inp) {
  const size_t i = (size_t)blockIdx.x * blockDim.x + threadIdx.x;
  if (i >= (size_t)CB * 8 * CD) return;
  const size_t b = i / ((size_t)8 * CD), r = i % ((size_t)8 * CD);
  const size_t j = r / CD, d = r % CD;
  o[i] = (bf16)(inp[((size_t)b * CS + 2 * j) * CD + d] +
                inp[((size_t)b * CS + 2 * j + 1) * CD + d]);
}
// non-first layers: update even positions of x from Hterm/Tterm
__global__ void k_mix_combine(float* __restrict__ X, const float* __restrict__ H,
                              const float* __restrict__ T) {
  const size_t i = (size_t)blockIdx.x * blockDim.x + threadIdx.x;
  if (i >= (size_t)CB * 9 * CD) return;
  const size_t b = i / ((size_t)9 * CD), r = i % ((size_t)9 * CD);
  const size_t e = r / CD, d = r % CD;
  const size_t xi = ((size_t)b * CS + 2 * e) * CD + d;
  if (e == 0)      X[xi] += CRW * H[((size_t)b * 8) * CD + d];
  else if (e == 8) X[xi] += CRW * T[((size_t)b * 8 + 7) * CD + d];
  else             X[xi] += 0.5f * CRW * T[((size_t)b * 8 + e - 1) * CD + d] +
                            0.5f * CRW * H[((size_t)b * 8 + e) * CD + d];
}
// first layer scan pieces
__global__ void k_mix_u0(float* __restrict__ U, float* __restrict__ X,
                         const float* __restrict__ H) {
  const size_t i = (size_t)blockIdx.x * blockDim.x + threadIdx.x;
  if (i >= (size_t)CB * CD) return;
  const size_t b = i / CD, d = i % CD;
  const float u = X[((size_t)b * CS) * CD + d] + CRW * H[((size_t)b * 8) * CD + d];
  U[i] = u;
  X[((size_t)b * CS) * CD + d] = u;
}
__global__ void k_build_uin(bf16* __restrict__ o, const float* __restrict__ U,
                            const float* __restrict__ X, int jj) {
  const size_t i = (size_t)blockIdx.x * blockDim.x + threadIdx.x;
  if (i >= (size_t)CB * CD) return;
  const size_t b = i / CD, d = i % CD;
  o[i] = (bf16)(U[i] + X[((size_t)b * CS + 2 * jj + 1) * CD + d]);
}
__global__ void k_scan_update(float* __restrict__ U, float* __restrict__ X,
                              const float* __restrict__ T, const float* __restrict__ H,
                              int jj) {
  const size_t i = (size_t)blockIdx.x * blockDim.x + threadIdx.x;
  if (i >= (size_t)CB * CD) return;
  const size_t b = i / CD, d = i % CD;
  const float h = (jj < 7) ? H[((size_t)b * 8 + jj + 1) * CD + d] : 0.f;
  const float ct = (jj == 7) ? CRW : 0.5f * CRW;
  const float ch = (jj == 7) ? 0.f : 0.5f * CRW;
  const float un = X[((size_t)b * CS + 2 * jj + 2) * CD + d] + ct * T[i] + ch * h;
  U[i] = un;
  X[((size_t)b * CS + 2 * jj + 2) * CD + d] = un;
}

// ---------------------------------------------------------------------------
// Host orchestration
// ---------------------------------------------------------------------------
constexpr size_t W_DD = (size_t)CD * CD;
constexpr size_t W_DF = (size_t)CD * CDFF;
constexpr size_t OFF_WQ = 0;
constexpr size_t OFF_WK = OFF_WQ + W_DD;
constexpr size_t OFF_WV = OFF_WK + W_DD;
constexpr size_t OFF_WO = OFF_WV + W_DD;
constexpr size_t OFF_W1 = OFF_WO + W_DD;
constexpr size_t OFF_W2 = OFF_W1 + W_DF;
constexpr size_t OFF_HW = OFF_W2 + W_DF;
constexpr size_t OFF_TW = OFF_HW + W_DD;
constexpr size_t LWE    = OFF_TW + W_DD;  // bf16 elems per layer

extern "C" void kernel_launch(void* const* d_in, const int* in_sizes, int n_in,
                              void* d_out, int out_size, void* d_ws, size_t ws_size,
                              hipStream_t stream) {
  (void)in_sizes; (void)n_in; (void)out_size; (void)ws_size;
  const float* xin   = (const float*)d_in[0];
  const int*   mask  = (const int*)d_in[1];
  const float* pWq   = (const float*)d_in[2];
  const float* pbq   = (const float*)d_in[3];
  const float* pWk   = (const float*)d_in[4];
  const float* pbk   = (const float*)d_in[5];
  const float* pWv   = (const float*)d_in[6];
  const float* pbv   = (const float*)d_in[7];
  const float* pWo   = (const float*)d_in[8];
  const float* pbo   = (const float*)d_in[9];
  const float* pln1g = (const float*)d_in[10];
  const float* pln1b = (const float*)d_in[11];
  const float* pln2g = (const float*)d_in[12];
  const float* pln2b = (const float*)d_in[13];
  const float* pW1   = (const float*)d_in[14];
  const float* pb1   = (const float*)d_in[15];
  const float* pW2   = (const float*)d_in[16];
  const float* pb2   = (const float*)d_in[17];
  const float* phW   = (const float*)d_in[18];
  const float* phb   = (const float*)d_in[19];
  const float* ptW   = (const float*)d_in[20];
  const float* ptb   = (const float*)d_in[21];
  const float* plnfg = (const float*)d_in[22];
  const float* plnfb = (const float*)d_in[23];

  constexpr size_t NT   = (size_t)CB * CS * CD;
  constexpr size_t NFF  = (size_t)CB * CS * CDFF;
  constexpr size_t NMIX = (size_t)CB * 8 * CD;
  constexpr size_t NBD  = (size_t)CB * CD;

  char* wsb = (char*)d_ws;
  size_t off = 0;
  auto alloc = [&](size_t bytes) -> void* {
    void* p = wsb + off;
    off += (bytes + 255) & ~(size_t)255;
    return p;
  };
  bf16*  WB    = (bf16*) alloc(LWE * CL * sizeof(bf16));
  float* X     = (float*)alloc(NT * 4);
  float* INP   = (float*)alloc(NT * 4);
  float* LNB   = (float*)alloc(NT * 4);
  float* Qb    = (float*)alloc(NT * 4);
  float* Kb    = (float*)alloc(NT * 4);
  float* Vb    = (float*)alloc(NT * 4);
  float* ATT   = (float*)alloc(NT * 4);
  bf16*  ABF   = (bf16*) alloc(NT * 2);
  bf16*  HBF   = (bf16*) alloc(NFF * 2);
  float* MIXH  = (float*)alloc(NMIX * 4);
  float* MIXT  = (float*)alloc(NMIX * 4);
  bf16*  MIXIN = (bf16*) alloc(NMIX * 2);
  float* Ub    = (float*)alloc(NBD * 4);

  const dim3 tb(256);
  // 1) fp32 weights -> transposed bf16 (every call: no cached state allowed)
  for (int i = 0; i < CL; ++i) {
    bf16* lw = WB + (size_t)i * LWE;
    k_wcvt<<<dim3(CD / 32, CD / 32), tb, 0, stream>>>(pWq + (size_t)i * W_DD, lw + OFF_WQ, CD, CD);
    k_wcvt<<<dim3(CD / 32, CD / 32), tb, 0, stream>>>(pWk + (size_t)i * W_DD, lw + OFF_WK, CD, CD);
    k_wcvt<<<dim3(CD / 32, CD / 32), tb, 0, stream>>>(pWv + (size_t)i * W_DD, lw + OFF_WV, CD, CD);
    k_wcvt<<<dim3(CD / 32, CD / 32), tb, 0, stream>>>(pWo + (size_t)i * W_DD, lw + OFF_WO, CD, CD);
    k_wcvt<<<dim3(CDFF / 32, CD / 32), tb, 0, stream>>>(pW1 + (size_t)i * W_DF, lw + OFF_W1, CD, CDFF);
    k_wcvt<<<dim3(CD / 32, CDFF / 32), tb, 0, stream>>>(pW2 + (size_t)i * W_DF, lw + OFF_W2, CDFF, CD);
    k_wcvt<<<dim3(CD / 32, CD / 32), tb, 0, stream>>>(phW + (size_t)i * W_DD, lw + OFF_HW, CD, CD);
    k_wcvt<<<dim3(CD / 32, CD / 32), tb, 0, stream>>>(ptW + (size_t)i * W_DD, lw + OFF_TW, CD, CD);
  }

  const int GNT  = (int)((NT + 255) / 256);
  const int GMIX = (int)((NMIX + 255) / 256);
  const int GBD  = (int)((NBD + 255) / 256);
  const int G9   = (int)(((size_t)CB * 9 * CD + 255) / 256);
  const int MR   = CB * CS;  // 8704 = 68 * 128

  k_copy<<<GNT, tb, 0, stream>>>(X, xin, NT);
  k_copy<<<GNT, tb, 0, stream>>>(INP, xin, NT);

  for (int i = 0; i < CL; ++i) {
    bf16* lw = WB + (size_t)i * LWE;
    // ---------------- mix ----------------
    if (i == 0) {
      k_build_hin<<<GMIX, tb, 0, stream>>>(MIXIN, INP);
      k_gemm<true, false, false><<<dim3(4096 / BM, CD / BN), tb, 0, stream>>>(
          MIXIN, lw + OFF_HW, phb, nullptr, MIXH, 4096, CD, CD);
      k_mix_u0<<<GBD, tb, 0, stream>>>(Ub, X, MIXH);
      for (int jj = 0; jj < 8; ++jj) {  // sequential scan (data dependent)
        k_build_uin<<<GBD, tb, 0, stream>>>(MIXIN, Ub, X, jj);
        k_gemm<true, false, false><<<dim3(CB / BM, CD / BN), tb, 0, stream>>>(
            MIXIN, lw + OFF_TW, ptb, nullptr, MIXT, CB, CD, CD);
        k_scan_update<<<GBD, tb, 0, stream>>>(Ub, X, MIXT, MIXH, jj);
      }
    } else {
      k_build_hin<<<GMIX, tb, 0, stream>>>(MIXIN, INP);
      k_gemm<true, false, false><<<dim3(4096 / BM, CD / BN), tb, 0, stream>>>(
          MIXIN, lw + OFF_HW, phb + (size_t)i * CD, nullptr, MIXH, 4096, CD, CD);
      k_build_tin<<<GMIX, tb, 0, stream>>>(MIXIN, INP);
      k_gemm<true, false, false><<<dim3(4096 / BM, CD / BN), tb, 0, stream>>>(
          MIXIN, lw + OFF_TW, ptb + (size_t)i * CD, nullptr, MIXT, 4096, CD, CD);
      k_mix_combine<<<G9, tb, 0, stream>>>(X, MIXH, MIXT);
    }
    k_copy<<<GNT, tb, 0, stream>>>(INP, X, NT);  // inp = x (post-mix)

    // ---------------- attention ----------------
    k_ln<<<MR, tb, 0, stream>>>(X, pln1g + (size_t)i * CD, pln1b + (size_t)i * CD, LNB);
    k_cvt<<<GNT, tb, 0, stream>>>(ABF, LNB, NT);
    k_gemm<false, false, false><<<dim3(MR / BM, CD / BN), tb, 0, stream>>>(
        ABF, lw + OFF_WQ, pbq + (size_t)i * CD, nullptr, Qb, MR, CD, CD);
    k_gemm<false, false, false><<<dim3(MR / BM, CD / BN), tb, 0, stream>>>(
        ABF, lw + OFF_WK, pbk + (size_t)i * CD, nullptr, Kb, MR, CD, CD);
    k_gemm<false, false, false><<<dim3(MR / BM, CD / BN), tb, 0, stream>>>(
        ABF, lw + OFF_WV, pbv + (size_t)i * CD, nullptr, Vb, MR, CD, CD);
    k_attn<<<CB * CNH, dim3(64), 0, stream>>>(Qb, Kb, Vb, mask, ATT);
    k_cvt<<<GNT, tb, 0, stream>>>(ABF, ATT, NT);
    k_gemm<false, false, true><<<dim3(MR / BM, CD / BN), tb, 0, stream>>>(
        ABF, lw + OFF_WO, pbo + (size_t)i * CD, X, X, MR, CD, CD);  // +residual

    // ---------------- FFN ----------------
    k_ln<<<MR, tb, 0, stream>>>(X, pln2g + (size_t)i * CD, pln2b + (size_t)i * CD, LNB);
    k_cvt<<<GNT, tb, 0, stream>>>(ABF, LNB, NT);
    k_gemm<true, true, false><<<dim3(MR / BM, CDFF / BN), tb, 0, stream>>>(
        ABF, lw + OFF_W1, pb1 + (size_t)i * CDFF, nullptr, HBF, MR, CDFF, CD);  // gelu, bf16 out
    k_gemm<false, false, true><<<dim3(MR / BM, CD / BN), tb, 0, stream>>>(
        HBF, lw + OFF_W2, pb2 + (size_t)i * CD, X, X, MR, CD, CDFF);  // +residual
  }

  // final LN straight into d_out
  k_ln<<<CB * CS, tb, 0, stream>>>(X, plnfg, plnfb, (float*)d_out);
}
